// Block_78280073937290
// MI455X (gfx1250) — compile-verified, matrix operand order checked by reading, not run on hardware
//
#include <hip/hip_runtime.h>
#include <hip/hip_bf16.h>
#include <math.h>

// ---------------- problem constants (match reference) ----------------
#define S_   2048
#define D_   2048
#define NH   16
#define NKV  4
#define HD_  128
#define NE   8
#define TOPK 2
#define MI_  1408
#define MS_  1408
#define GG   12
#define CAP_ 640

typedef __bf16 bf16_t;
typedef __attribute__((ext_vector_type(16))) __bf16 v16bf;
typedef __attribute__((ext_vector_type(8)))  float  v8f;
typedef __attribute__((ext_vector_type(4)))  unsigned int u32x4;
typedef __attribute__((ext_vector_type(8)))  int i32x8;
typedef __attribute__((ext_vector_type(4)))  int i32x4;

static __device__ __forceinline__ v8f v8f_zero() {
  v8f z = {0.f,0.f,0.f,0.f,0.f,0.f,0.f,0.f};
  return z;
}

// =====================================================================
// Tensor Data Mover: async 2D bf16 tile -> LDS with row padding.
// tensor dims == tile dims (tiles never OOB here). data_size=1 (2B).
// pad: every (2<<padInterval) dwords of stored data, insert
// (padAmount+1) dwords -> gives padded LDS row strides for bank spread.
// =====================================================================
__device__ __forceinline__ void tdm_load_2d(
    unsigned ldsAddr, const void* gsrc,
    int tileD0, int tileD1, long strideElems,
    int padInterval, int padAmount)
{
  const unsigned long long ga = (unsigned long long)gsrc;
  u32x4 g0;
  g0.x = 1u;                                  // count=1, user, no gather
  g0.y = ldsAddr;                             // lds_addr [63:32]
  g0.z = (unsigned)ga;                        // global_addr [95:64]
  g0.w = (unsigned)((ga >> 32) & 0x1ffffffull) | (2u << 30); // [120:96], type=2
  i32x8 g1;
  g1[0] = (1 << 16) | (1 << 20) | (padInterval << 22) | (padAmount << 25);
  g1[1] = (int)((unsigned)tileD0 << 16);      // tensor_dim0 lo16 -> [63:48]
  g1[2] = (int)(((unsigned)tileD0 >> 16) | ((unsigned)tileD1 << 16));
  g1[3] = (int)((unsigned)tileD0 << 16);      // tensor_dim1 hi16=0 | tile_dim0
  g1[4] = tileD1;                             // tile_dim1 (tile_dim2=0)
  g1[5] = (int)(unsigned)(strideElems & 0xffffffff);
  g1[6] = (int)(unsigned)((strideElems >> 32) & 0xffff);
  g1[7] = 0;
  i32x4 zz4 = {0, 0, 0, 0};
  i32x8 zz8 = {0, 0, 0, 0, 0, 0, 0, 0};
  __builtin_amdgcn_tensor_load_to_lds(g0, g1, zz4, zz4, zz8, 0);
}

static __device__ __forceinline__ unsigned lds_off(const void* p) {
  return (unsigned)(unsigned long long)p; // LDS generic addr low 32 = DS offset
}

// =====================================================================
// bf16 WMMA GEMM: C[z] = A[z] @ B^T[z] (+resid). A,B already bf16,
// B pre-transposed to [N][K]. TDM double-buffered tile staging.
// =====================================================================
template <bool GATHER, bool RESID, bool OUTBF>
__global__ __launch_bounds__(256) void wmma_gemm_kernel(
    const bf16_t* __restrict__ A, int lda, long aZ,
    const bf16_t* __restrict__ Bt, int ldb, long bZ,
    void* __restrict__ Cv, int ldc, long cZ,
    const float* __restrict__ resid,
    const int* __restrict__ row_map,
    int M, int N, int K)
{
  constexpr int LDT = 40; // 32 bf16 data + 4 dword pad = 80B stride
  __shared__ bf16_t As[2][128 * LDT];
  __shared__ bf16_t Bs[2][128 * LDT];

  const int z = blockIdx.z;
  A  += (long)z * aZ;
  Bt += (long)z * bZ;
  const int* rmap = GATHER ? (row_map + (long)z * M) : nullptr;

  const int bm0 = blockIdx.y * 128;
  const int bn0 = blockIdx.x * 128;
  const int tid  = threadIdx.x;
  const int lane = tid & 31;
  const int wid  = tid >> 5;
  const int half = lane >> 4;
  const int l15  = lane & 15;
  const int wm = (wid & 3) * 32;
  const int wn = (wid >> 2) * 64;

  v8f acc[2][4];
  #pragma unroll
  for (int i = 0; i < 2; ++i)
    #pragma unroll
    for (int j = 0; j < 4; ++j) acc[i][j] = v8f_zero();

  // preload tile 0
  if (wid == 0) {
    tdm_load_2d(lds_off(&Bs[0][0]), Bt + (long)bn0 * ldb, 32, 128, ldb, 3, 3);
    if (!GATHER)
      tdm_load_2d(lds_off(&As[0][0]), A + (long)bm0 * lda, 32, 128, lda, 3, 3);
  }

  int b = 0;
  for (int k0 = 0; k0 < K; k0 += 32) {
    if (GATHER) {
      // 128 rows x 64B, already bf16: 512 16B chunks / 256 threads
      #pragma unroll
      for (int p = 0; p < 2; ++p) {
        const int c = tid + p * 256;
        const int row = c >> 2, kc = c & 3;
        const int src = rmap[bm0 + row];
        uint4 v = make_uint4(0u, 0u, 0u, 0u);
        if (src >= 0)
          v = *reinterpret_cast<const uint4*>(A + (long)src * lda + k0 + kc * 8);
        *reinterpret_cast<uint4*>(&As[b][row * LDT + kc * 8]) = v;
      }
    }
    if (wid == 0) __builtin_amdgcn_s_wait_tensorcnt(0);
    __syncthreads(); // tile k0 resident in buffer b
    if (wid == 0 && k0 + 32 < K) {
      tdm_load_2d(lds_off(&Bs[b ^ 1][0]), Bt + (long)bn0 * ldb + (k0 + 32),
                  32, 128, ldb, 3, 3);
      if (!GATHER)
        tdm_load_2d(lds_off(&As[b ^ 1][0]), A + (long)bm0 * lda + (k0 + 32),
                    32, 128, lda, 3, 3);
    }

    v16bf afrag[2], bfrag[4];
    #pragma unroll
    for (int i = 0; i < 2; ++i) {
      const bf16_t* p = &As[b][(wm + i * 16 + l15) * LDT + half * 8];
      #pragma unroll
      for (int e = 0; e < 8; ++e) { afrag[i][e] = p[e]; afrag[i][e + 8] = p[e + 16]; }
    }
    #pragma unroll
    for (int j = 0; j < 4; ++j) {
      const bf16_t* p = &Bs[b][(wn + j * 16 + l15) * LDT + half * 8];
      #pragma unroll
      for (int e = 0; e < 8; ++e) { bfrag[j][e] = p[e]; bfrag[j][e + 8] = p[e + 16]; }
    }
    #pragma unroll
    for (int i = 0; i < 2; ++i)
      #pragma unroll
      for (int j = 0; j < 4; ++j)
        acc[i][j] = __builtin_amdgcn_wmma_f32_16x16x32_bf16(
            false, afrag[i], false, bfrag[j], (short)0, acc[i][j], false, false);
    __syncthreads(); // all reads of buffer b complete
    b ^= 1;
  }

  // epilogue: row = r + 8*half, col = lane&15
  #pragma unroll
  for (int i = 0; i < 2; ++i)
    #pragma unroll
    for (int j = 0; j < 4; ++j)
      #pragma unroll
      for (int r = 0; r < 8; ++r) {
        const int row = bm0 + wm + i * 16 + half * 8 + r;
        const int col = bn0 + wn + j * 16 + l15;
        const long idx = (long)z * cZ + (long)row * ldc + col;
        float vv = acc[i][j][r];
        if (RESID) vv += resid[idx];
        if (OUTBF) ((bf16_t*)Cv)[idx] = (bf16_t)vv;
        else       ((float*)Cv)[idx]  = vv;
      }
}

// =====================================================================
// Flash attention, causal, GQA; bf16 q/k/vT in, bf16 gated o out.
// K/V tiles streamed by TDM, double-buffered.
// K LDS rows: 128 elems with 8-elem gap at d=64 (stride 144).
// =====================================================================
__global__ __launch_bounds__(256) void attn_kernel(
    const bf16_t* __restrict__ qb, const bf16_t* __restrict__ kb,
    const bf16_t* __restrict__ vt, const float* __restrict__ gatev,
    bf16_t* __restrict__ ob)
{
  __shared__ bf16_t Ks[2][32 * 144];  // [key][d-with-gap]
  __shared__ bf16_t Vs[2][128 * 40];  // [d][key]
  __shared__ bf16_t Ps[8 * 16 * 40];  // per-wave P transpose buffer

  const int qblk = blockIdx.x;
  const int head = blockIdx.y;
  const int kvh  = head >> 2;
  const int tid  = threadIdx.x;
  const int lane = tid & 31;
  const int wid  = tid >> 5;
  const int half = lane >> 4;
  const int l15  = lane & 15;
  const int q0w  = qblk * 128 + wid * 16;
  const float scale = 0.08838834764831845f; // 1/sqrt(128)

  // q fragments (scale folded)
  v16bf qa[4];
  {
    const bf16_t* qp = qb + (long)(q0w + l15) * (NH * HD_) + head * HD_;
    #pragma unroll
    for (int dc = 0; dc < 4; ++dc) {
      const int base = dc * 32 + half * 8;
      #pragma unroll
      for (int e = 0; e < 8; ++e) {
        qa[dc][e]     = (bf16_t)((float)qp[base + e] * scale);
        qa[dc][e + 8] = (bf16_t)((float)qp[base + 16 + e] * scale);
      }
    }
  }

  v8f oacc[8];
  #pragma unroll
  for (int dc = 0; dc < 8; ++dc) oacc[dc] = v8f_zero();
  float mrow[8], lrow[8];
  #pragma unroll
  for (int r = 0; r < 8; ++r) { mrow[r] = -1e30f; lrow[r] = 0.f; }

  const int kend = qblk * 128 + 128;
  if (wid == 0) {
    tdm_load_2d(lds_off(&Ks[0][0]), kb + kvh * HD_, 128, 32, NKV * HD_, 4, 3);
    tdm_load_2d(lds_off(&Vs[0][0]), vt + (long)(kvh * HD_) * S_, 32, 128, S_, 3, 3);
  }

  int b = 0;
  for (int kb0 = 0; kb0 < kend; kb0 += 32) {
    if (wid == 0) __builtin_amdgcn_s_wait_tensorcnt(0);
    __syncthreads();
    if (wid == 0 && kb0 + 32 < kend) {
      tdm_load_2d(lds_off(&Ks[b ^ 1][0]), kb + (long)(kb0 + 32) * (NKV * HD_) + kvh * HD_,
                  128, 32, NKV * HD_, 4, 3);
      tdm_load_2d(lds_off(&Vs[b ^ 1][0]), vt + (long)(kvh * HD_) * S_ + (kb0 + 32),
                  32, 128, S_, 3, 3);
    }

    if (kb0 <= q0w + 15) { // wave-uniform causal skip
      v8f sacc[2];
      sacc[0] = v8f_zero(); sacc[1] = v8f_zero();
      #pragma unroll
      for (int sub = 0; sub < 2; ++sub)
        #pragma unroll
        for (int dc = 0; dc < 4; ++dc) {
          const int d0 = dc * 32 + half * 8;
          const int d1 = d0 + 16;
          const bf16_t* p0 = &Ks[b][(sub * 16 + l15) * 144 + d0 + (d0 >= 64 ? 8 : 0)];
          const bf16_t* p1 = &Ks[b][(sub * 16 + l15) * 144 + d1 + (d1 >= 64 ? 8 : 0)];
          v16bf kf;
          #pragma unroll
          for (int e = 0; e < 8; ++e) { kf[e] = p0[e]; kf[e + 8] = p1[e]; }
          sacc[sub] = __builtin_amdgcn_wmma_f32_16x16x32_bf16(
              false, qa[dc], false, kf, (short)0, sacc[sub], false, false);
        }

      #pragma unroll
      for (int r = 0; r < 8; ++r) {
        const int qidx = q0w + half * 8 + r;
        const bool m0 = (kb0 + l15) > qidx;
        const bool m1 = (kb0 + 16 + l15) > qidx;
        float s0 = m0 ? -1e30f : sacc[0][r];
        float s1 = m1 ? -1e30f : sacc[1][r];
        float v = fmaxf(s0, s1);
        #pragma unroll
        for (int off = 1; off < 16; off <<= 1) v = fmaxf(v, __shfl_xor(v, off, 16));
        const float mn = fmaxf(mrow[r], v);
        float p0 = m0 ? 0.f : __expf(s0 - mn);
        float p1 = m1 ? 0.f : __expf(s1 - mn);
        float rs = p0 + p1;
        #pragma unroll
        for (int off = 1; off < 16; off <<= 1) rs += __shfl_xor(rs, off, 16);
        const float sc = __expf(mrow[r] - mn);
        mrow[r] = mn;
        lrow[r] = lrow[r] * sc + rs;
        #pragma unroll
        for (int dc = 0; dc < 8; ++dc) oacc[dc][r] *= sc;
        bf16_t* pw = &Ps[wid * 640 + (half * 8 + r) * 40 + l15];
        pw[0]  = (bf16_t)p0;
        pw[16] = (bf16_t)p1;
      }
      asm volatile("s_wait_dscnt 0" ::: "memory"); // per-wave P handoff

      v16bf pf;
      {
        const bf16_t* p = &Ps[wid * 640 + l15 * 40 + half * 8];
        #pragma unroll
        for (int e = 0; e < 8; ++e) { pf[e] = p[e]; pf[e + 8] = p[e + 16]; }
      }
      #pragma unroll
      for (int dc = 0; dc < 8; ++dc) {
        v16bf vf;
        const bf16_t* p = &Vs[b][(dc * 16 + l15) * 40 + half * 8];
        #pragma unroll
        for (int e = 0; e < 8; ++e) { vf[e] = p[e]; vf[e + 8] = p[e + 16]; }
        oacc[dc] = __builtin_amdgcn_wmma_f32_16x16x32_bf16(
            false, pf, false, vf, (short)0, oacc[dc], false, false);
      }
    }
    __syncthreads();
    b ^= 1;
  }

  #pragma unroll
  for (int r = 0; r < 8; ++r) {
    const int srow = q0w + half * 8 + r;
    const float linv = (lrow[r] > 0.f) ? (1.f / lrow[r]) : 0.f;
    const float g = gatev[(long)srow * NH + head];
    #pragma unroll
    for (int dc = 0; dc < 8; ++dc)
      ob[(long)srow * (NH * HD_) + head * HD_ + dc * 16 + l15] =
          (bf16_t)(oacc[dc][r] * linv * g);
  }
}

// =====================================================================
// One-time weight transpose + f32->bf16 convert: Bt[n][k] = W[k][n]
// =====================================================================
__global__ __launch_bounds__(256) void transpose_convert_kernel(
    const float* __restrict__ W, bf16_t* __restrict__ Bt, int K, int N)
{
  __shared__ float tile[32][33];
  const long z = blockIdx.z;
  W  += z * (long)K * N;
  Bt += z * (long)K * N;
  const int k0 = blockIdx.y * 32, n0 = blockIdx.x * 32;
  const int tj = threadIdx.x & 31, ti = threadIdx.x >> 5;
  #pragma unroll
  for (int p = 0; p < 4; ++p)
    tile[ti + p * 8][tj] = W[(long)(k0 + ti + p * 8) * N + n0 + tj];
  __syncthreads();
  #pragma unroll
  for (int p = 0; p < 4; ++p)
    Bt[(long)(n0 + ti + p * 8) * K + k0 + tj] = (bf16_t)tile[tj][ti + p * 8];
}

__global__ __launch_bounds__(256) void transpose_bf16_kernel(
    const bf16_t* __restrict__ in, bf16_t* __restrict__ out, int R, int C)
{
  __shared__ bf16_t tile[32][33];
  const int r0 = blockIdx.y * 32, c0 = blockIdx.x * 32;
  const int tj = threadIdx.x & 31, ti = threadIdx.x >> 5;
  #pragma unroll
  for (int p = 0; p < 4; ++p)
    tile[ti + p * 8][tj] = in[(long)(r0 + ti + p * 8) * C + c0 + tj];
  __syncthreads();
  #pragma unroll
  for (int p = 0; p < 4; ++p)
    out[(long)(c0 + ti + p * 8) * R + r0 + tj] = tile[tj][ti + p * 8];
}

// =====================================================================
// Small / elementwise kernels
// =====================================================================
__global__ __launch_bounds__(256) void rmsnorm_bf16_kernel(
    const float* __restrict__ x, const float* __restrict__ w, bf16_t* __restrict__ out)
{
  __shared__ float red[8];
  const int s = blockIdx.x, tid = threadIdx.x;
  const float* xr = x + (long)s * D_;
  float acc = 0.f;
  for (int d = tid; d < D_; d += 256) { float v = xr[d]; acc += v * v; }
  #pragma unroll
  for (int off = 1; off < 32; off <<= 1) acc += __shfl_xor(acc, off, 32);
  if ((tid & 31) == 0) red[tid >> 5] = acc;
  __syncthreads();
  float tot = 0.f;
  #pragma unroll
  for (int i = 0; i < 8; ++i) tot += red[i];
  const float r = rsqrtf(tot / (float)D_ + 1e-5f);
  bf16_t* orow = out + (long)s * D_;
  for (int d = tid; d < D_; d += 256) orow[d] = (bf16_t)(xr[d] * r * w[d]);
}

// qk-norm (eps 1e-6) + RoPE; f32 in (from GEMM), bf16 out; wave per (s, head)
__global__ __launch_bounds__(256) void rope_qknorm_kernel(
    const float* __restrict__ q, const float* __restrict__ k,
    bf16_t* __restrict__ qb, bf16_t* __restrict__ kb)
{
  const int tid = threadIdx.x, lane = tid & 31, wid = tid >> 5;
  const int gw = blockIdx.x * 8 + wid;
  const int s = gw / (NH + NKV);
  const int h = gw % (NH + NKV);
  const float* p;
  bf16_t* o;
  if (h < NH) { p = q + (long)s * (NH * HD_) + h * HD_;  o = qb + (long)s * (NH * HD_) + h * HD_; }
  else        { p = k + (long)s * (NKV * HD_) + (h - NH) * HD_;
                o = kb + (long)s * (NKV * HD_) + (h - NH) * HD_; }
  float a = p[lane], bb = p[lane + 32], c0 = p[lane + 64], c1 = p[lane + 96];
  float ss = a * a + bb * bb + c0 * c0 + c1 * c1;
  #pragma unroll
  for (int off = 1; off < 32; off <<= 1) ss += __shfl_xor(ss, off, 32);
  const float r = rsqrtf(ss / (float)HD_ + 1e-6f);
  a *= r; bb *= r; c0 *= r; c1 *= r;
  const float inv = __expf(-(float)lane * (logf(1024.f) / 32.f));
  const float ang = (float)s * inv;
  float sn, cs; __sincosf(ang, &sn, &cs);
  o[lane]      = (bf16_t)(a * cs - bb * sn);
  o[lane + 32] = (bf16_t)(bb * cs + a * sn);
  o[lane + 64] = (bf16_t)c0;
  o[lane + 96] = (bf16_t)c1;
}

__global__ void gate_kernel(const bf16_t* __restrict__ xn, const float* __restrict__ ag,
                            float* __restrict__ gatev)
{
  const int i = blockIdx.x * blockDim.x + threadIdx.x;
  if (i >= S_ * NH) return;
  const int s = i / NH, n = i % NH;
  float acc = 0.f;
  #pragma unroll
  for (int g = 0; g < GG; ++g) acc += (float)xn[(long)s * D_ + g] * ag[g * NH + n];
  gatev[i] = 2.f / (1.f + __expf(-acc));
}

__global__ void router_kernel(const bf16_t* __restrict__ u, const float* __restrict__ wr,
                              float* __restrict__ probs, float* __restrict__ zl)
{
  const int t = blockIdx.x * blockDim.x + threadIdx.x;
  if (t >= S_) return;
  float acc[NE];
  #pragma unroll
  for (int e = 0; e < NE; ++e) acc[e] = 0.f;
  const bf16_t* ur = u + (long)t * D_;
  for (int d = 0; d < D_; ++d) {
    const float xv = (float)ur[d];
    const float* w = wr + d * NE;
    #pragma unroll
    for (int e = 0; e < NE; ++e) acc[e] += xv * w[e];
  }
  float mx = acc[0];
  #pragma unroll
  for (int e = 1; e < NE; ++e) mx = fmaxf(mx, acc[e]);
  float pr[NE], sm = 0.f;
  #pragma unroll
  for (int e = 0; e < NE; ++e) { pr[e] = __expf(acc[e] - mx); sm += pr[e]; }
  #pragma unroll
  for (int e = 0; e < NE; ++e) probs[(long)t * NE + e] = pr[e] / sm;
  zl[t] = mx + logf(sm);
}

// Deterministic top-2 + capacity assignment (stable argsort order)
__global__ __launch_bounds__(256) void route_build_kernel(
    const float* __restrict__ probs, int* __restrict__ top_i, float* __restrict__ top_w,
    int* __restrict__ row_map, int* __restrict__ slot_t, int* __restrict__ counts)
{
  const int tid = threadIdx.x;
  for (int t = tid; t < S_; t += 256) {
    const float* p = probs + (long)t * NE;
    int i1 = 0; float p1 = p[0];
    #pragma unroll
    for (int e = 1; e < NE; ++e) if (p[e] > p1) { p1 = p[e]; i1 = e; }
    int i2 = -1; float p2 = -1.f;
    #pragma unroll
    for (int e = 0; e < NE; ++e) if (e != i1 && p[e] > p2) { p2 = p[e]; i2 = e; }
    const float wsum = p1 + p2;
    top_i[2 * t] = i1; top_i[2 * t + 1] = i2;
    top_w[2 * t] = p1 / wsum; top_w[2 * t + 1] = p2 / wsum;
  }
  for (int i = tid; i < NE * CAP_; i += 256) row_map[i] = -1;
  __syncthreads();
  if (tid < NE) {
    const int e = tid;
    int c = 0;
    for (int idx = 0; idx < S_ * TOPK; ++idx) {
      if (top_i[idx] == e) {
        if (c < CAP_) { row_map[e * CAP_ + c] = idx >> 1; slot_t[idx] = e * CAP_ + c; }
        else          { slot_t[idx] = -1; }
        ++c;
      }
    }
    counts[e] = c;
  }
}

__global__ void silu_mul_bf_kernel(const bf16_t* __restrict__ g, const bf16_t* __restrict__ u,
                                   bf16_t* __restrict__ out, long n)
{
  const long i = (long)blockIdx.x * blockDim.x + threadIdx.x;
  if (i >= n) return;
  const float x = (float)g[i];
  out[i] = (bf16_t)((x / (1.f + __expf(-x))) * (float)u[i]);
}

__global__ void moe_gather_kernel(const float* __restrict__ yb, const int* __restrict__ slot_t,
                                  const float* __restrict__ top_w, float* __restrict__ ymoe)
{
  const long i = (long)blockIdx.x * blockDim.x + threadIdx.x;
  if (i >= (long)S_ * D_) return;
  const int t = (int)(i / D_), d = (int)(i % D_);
  float acc = 0.f;
  #pragma unroll
  for (int j = 0; j < TOPK; ++j) {
    const int sl = slot_t[2 * t + j];
    if (sl >= 0) acc += top_w[2 * t + j] * yb[(long)sl * D_ + d];
  }
  ymoe[i] = acc;
}

__global__ void final_add_kernel(const float* __restrict__ h, const float* __restrict__ ymoe,
                                 const float* __restrict__ sd, float* __restrict__ out)
{
  const long i = (long)blockIdx.x * blockDim.x + threadIdx.x;
  if (i >= (long)S_ * D_) return;
  out[i] = h[i] + ymoe[i] + sd[i];
}

__global__ void aux_kernel(const float* __restrict__ probs, const float* __restrict__ zl,
                           const int* __restrict__ counts, float* __restrict__ outAux)
{
  __shared__ float ps[NE];
  __shared__ float z2s;
  const int tid = threadIdx.x;
  if (tid < NE) {
    float a = 0.f;
    for (int t = 0; t < S_; ++t) a += probs[(long)t * NE + tid];
    ps[tid] = a;
  }
  if (tid == NE) {
    float a = 0.f;
    for (int t = 0; t < S_; ++t) { const float z = zl[t]; a += z * z; }
    z2s = a;
  }
  __syncthreads();
  if (tid == 0) {
    float lb = 0.f;
    for (int e = 0; e < NE; ++e) {
      const float frac = (float)counts[e] / (float)(S_ * TOPK);
      lb += frac * (float)TOPK * (ps[e] / (float)S_);
    }
    lb *= (float)NE;
    outAux[0] = 0.01f * lb + 0.001f * (z2s / (float)S_);
  }
}

// =====================================================================
// Host driver. Workspace carved with 256B-aligned bump allocator
// (needs ~380 MB of d_ws).
// =====================================================================
extern "C" void kernel_launch(void* const* d_in, const int* in_sizes, int n_in,
                              void* d_out, int out_size, void* d_ws, size_t ws_size,
                              hipStream_t stream)
{
  (void)in_sizes; (void)n_in; (void)out_size; (void)ws_size;
  const float* x    = (const float*)d_in[0];
  const float* ln1  = (const float*)d_in[1];
  const float* ln2  = (const float*)d_in[2];
  const float* w_q  = (const float*)d_in[3];
  const float* w_k  = (const float*)d_in[4];
  const float* w_v  = (const float*)d_in[5];
  const float* w_o  = (const float*)d_in[6];
  const float* agw  = (const float*)d_in[7];
  const float* w_r  = (const float*)d_in[8];
  const float* wge  = (const float*)d_in[9];
  const float* wue  = (const float*)d_in[10];
  const float* wde  = (const float*)d_in[11];
  const float* wgs  = (const float*)d_in[12];
  const float* wus  = (const float*)d_in[13];
  const float* wds  = (const float*)d_in[14];
  float* out = (float*)d_out;

  char* W = (char*)d_ws;
  auto allocF = [&](long n) { float*  p = (float*)W;  W += ((n * 4 + 255) & ~255L); return p; };
  auto allocB = [&](long n) { bf16_t* p = (bf16_t*)W; W += ((n * 2 + 255) & ~255L); return p; };
  auto allocI = [&](long n) { int*    p = (int*)W;    W += ((n * 4 + 255) & ~255L); return p; };

  const long SD = (long)S_ * D_;
  // f32 buffers
  float* qf   = allocF(SD);
  float* kf   = allocF((long)S_ * NKV * HD_);
  float* h    = allocF(SD);
  float* yb   = allocF((long)NE * CAP_ * D_);
  float* ymoe = allocF(SD);
  float* sd   = allocF(SD);
  float* gatev= allocF((long)S_ * NH);
  float* probs= allocF((long)S_ * NE);
  float* zl   = allocF(S_);
  float* tw   = allocF((long)S_ * TOPK);
  // bf16 activations
  bf16_t* xnb = allocB(SD);
  bf16_t* qb  = allocB(SD);
  bf16_t* kbb = allocB((long)S_ * NKV * HD_);
  bf16_t* vb  = allocB((long)S_ * NKV * HD_);
  bf16_t* vt  = allocB((long)S_ * NKV * HD_);
  bf16_t* obb = allocB(SD);
  bf16_t* ub  = allocB(SD);
  bf16_t* gbb = allocB((long)NE * CAP_ * MI_);
  bf16_t* ubb = allocB((long)NE * CAP_ * MI_);
  bf16_t* heb = allocB((long)NE * CAP_ * MI_);
  bf16_t* gsb = allocB((long)S_ * MS_);
  bf16_t* usb = allocB((long)S_ * MS_);
  bf16_t* hsb = allocB((long)S_ * MS_);
  // bf16 transposed weights [N][K]
  bf16_t* wqT = allocB((long)D_ * D_);
  bf16_t* wkT = allocB((long)D_ * NKV * HD_);
  bf16_t* wvT = allocB((long)D_ * NKV * HD_);
  bf16_t* woT = allocB((long)D_ * D_);
  bf16_t* wgeT= allocB((long)NE * D_ * MI_);
  bf16_t* wueT= allocB((long)NE * D_ * MI_);
  bf16_t* wdeT= allocB((long)NE * D_ * MI_);
  bf16_t* wgsT= allocB((long)D_ * MS_);
  bf16_t* wusT= allocB((long)D_ * MS_);
  bf16_t* wdsT= allocB((long)D_ * MS_);
  // ints
  int* top_i   = allocI((long)S_ * TOPK);
  int* row_map = allocI((long)NE * CAP_);
  int* slot_t  = allocI((long)S_ * TOPK);
  int* counts  = allocI(NE);

  // ---- 0) weight transpose + convert (one bandwidth pass) ----------
  transpose_convert_kernel<<<dim3(64, 64, 1), 256, 0, stream>>>(w_q, wqT, D_, D_);
  transpose_convert_kernel<<<dim3(16, 64, 1), 256, 0, stream>>>(w_k, wkT, D_, NKV * HD_);
  transpose_convert_kernel<<<dim3(16, 64, 1), 256, 0, stream>>>(w_v, wvT, D_, NKV * HD_);
  transpose_convert_kernel<<<dim3(64, 64, 1), 256, 0, stream>>>(w_o, woT, D_, D_);
  transpose_convert_kernel<<<dim3(44, 64, NE), 256, 0, stream>>>(wge, wgeT, D_, MI_);
  transpose_convert_kernel<<<dim3(44, 64, NE), 256, 0, stream>>>(wue, wueT, D_, MI_);
  transpose_convert_kernel<<<dim3(64, 44, NE), 256, 0, stream>>>(wde, wdeT, MI_, D_);
  transpose_convert_kernel<<<dim3(44, 64, 1), 256, 0, stream>>>(wgs, wgsT, D_, MS_);
  transpose_convert_kernel<<<dim3(44, 64, 1), 256, 0, stream>>>(wus, wusT, D_, MS_);
  transpose_convert_kernel<<<dim3(64, 44, 1), 256, 0, stream>>>(wds, wdsT, MS_, D_);

  // ---- 1) xn = rmsnorm(x, ln1) (bf16) ------------------------------
  rmsnorm_bf16_kernel<<<S_, 256, 0, stream>>>(x, ln1, xnb);

  // ---- 2) q/k/v projections ----------------------------------------
  wmma_gemm_kernel<false, false, false><<<dim3(16, 16, 1), 256, 0, stream>>>(
      xnb, D_, 0, wqT, D_, 0, qf, D_, 0, nullptr, nullptr, S_, D_, D_);
  wmma_gemm_kernel<false, false, false><<<dim3(4, 16, 1), 256, 0, stream>>>(
      xnb, D_, 0, wkT, D_, 0, kf, NKV * HD_, 0, nullptr, nullptr, S_, NKV * HD_, D_);
  wmma_gemm_kernel<false, false, true><<<dim3(4, 16, 1), 256, 0, stream>>>(
      xnb, D_, 0, wvT, D_, 0, vb, NKV * HD_, 0, nullptr, nullptr, S_, NKV * HD_, D_);

  // ---- 3) qk-norm + rope (f32 -> bf16), v transpose, gate ----------
  rope_qknorm_kernel<<<(S_ * (NH + NKV)) / 8, 256, 0, stream>>>(qf, kf, qb, kbb);
  transpose_bf16_kernel<<<dim3(16, 64, 1), 256, 0, stream>>>(vb, vt, S_, NKV * HD_);
  gate_kernel<<<(S_ * NH + 255) / 256, 256, 0, stream>>>(xnb, agw, gatev);

  // ---- 4) flash attention (TDM-staged K/V) -> gated bf16 o ---------
  attn_kernel<<<dim3(S_ / 128, NH, 1), 256, 0, stream>>>(qb, kbb, vt, gatev, obb);

  // ---- 5) h = x + o @ w_o ------------------------------------------
  wmma_gemm_kernel<false, true, false><<<dim3(16, 16, 1), 256, 0, stream>>>(
      obb, D_, 0, woT, D_, 0, h, D_, 0, x, nullptr, S_, D_, D_);

  // ---- 6) u = rmsnorm(h, ln2) (bf16) -------------------------------
  rmsnorm_bf16_kernel<<<S_, 256, 0, stream>>>(h, ln2, ub);

  // ---- 7) router + deterministic routing ---------------------------
  router_kernel<<<S_ / 256, 256, 0, stream>>>(ub, w_r, probs, zl);
  route_build_kernel<<<1, 256, 0, stream>>>(probs, top_i, tw, row_map, slot_t, counts);

  // ---- 8) MoE expert GEMMs (token gather on the fly) ---------------
  wmma_gemm_kernel<true, false, true><<<dim3(MI_ / 128, CAP_ / 128, NE), 256, 0, stream>>>(
      ub, D_, 0, wgeT, D_, (long)D_ * MI_, gbb, MI_, (long)CAP_ * MI_,
      nullptr, row_map, CAP_, MI_, D_);
  wmma_gemm_kernel<true, false, true><<<dim3(MI_ / 128, CAP_ / 128, NE), 256, 0, stream>>>(
      ub, D_, 0, wueT, D_, (long)D_ * MI_, ubb, MI_, (long)CAP_ * MI_,
      nullptr, row_map, CAP_, MI_, D_);
  {
    const long n = (long)NE * CAP_ * MI_;
    silu_mul_bf_kernel<<<(unsigned)((n + 255) / 256), 256, 0, stream>>>(gbb, ubb, heb, n);
  }
  wmma_gemm_kernel<false, false, false><<<dim3(D_ / 128, CAP_ / 128, NE), 256, 0, stream>>>(
      heb, MI_, (long)CAP_ * MI_, wdeT, MI_, (long)D_ * MI_,
      yb, D_, (long)CAP_ * D_, nullptr, nullptr, CAP_, D_, MI_);
  moe_gather_kernel<<<(unsigned)((SD + 255) / 256), 256, 0, stream>>>(yb, slot_t, tw, ymoe);

  // ---- 9) shared FFN ----------------------------------------------
  wmma_gemm_kernel<false, false, true><<<dim3(MS_ / 128, 16, 1), 256, 0, stream>>>(
      ub, D_, 0, wgsT, D_, 0, gsb, MS_, 0, nullptr, nullptr, S_, MS_, D_);
  wmma_gemm_kernel<false, false, true><<<dim3(MS_ / 128, 16, 1), 256, 0, stream>>>(
      ub, D_, 0, wusT, D_, 0, usb, MS_, 0, nullptr, nullptr, S_, MS_, D_);
  {
    const long n = (long)S_ * MS_;
    silu_mul_bf_kernel<<<(unsigned)((n + 255) / 256), 256, 0, stream>>>(gsb, usb, hsb, n);
  }
  wmma_gemm_kernel<false, false, false><<<dim3(16, 16, 1), 256, 0, stream>>>(
      hsb, MS_, 0, wdsT, MS_, 0, sd, D_, 0, nullptr, nullptr, S_, D_, MS_);

  // ---- 10) output + aux -------------------------------------------
  final_add_kernel<<<(unsigned)((SD + 255) / 256), 256, 0, stream>>>(h, ymoe, sd, out);
  aux_kernel<<<1, 64, 0, stream>>>(probs, zl, counts, out + SD);
}
